// DeepseekV2MoE_4664334484017
// MI455X (gfx1250) — matrix operand order repaired and checked by reading, not compiled, not run on hardware
//
#include <hip/hip_runtime.h>

// ---------------- problem constants ----------------
#define HDIM   2048   // hidden size
#define IDIM   1408   // routed expert intermediate
#define NEXP   16
#define NTOK   2048   // B*S tokens
#define SIDIM  2816   // shared expert intermediate (2*IDIM)
#define RSCALE 16.0f

// ---------------- types ----------------
typedef __bf16        v16bf __attribute__((ext_vector_type(16)));
typedef float         v8f   __attribute__((ext_vector_type(8)));
typedef unsigned int  u32x4 __attribute__((ext_vector_type(4)));
typedef float         f32x4 __attribute__((ext_vector_type(4)));

union FragB16 { u32x4 q[2]; v16bf v; };   // 32 bytes: one WMMA bf16 operand
union R16     { f32x4 v[4]; float f[16]; };

// LDS tile: 128 rows x 32 bf16, row stride 80 B (64 data + 16 pad, keeps 16B align)
#define LROW 80
#define TILE_BYTES (128 * LROW)

__device__ __forceinline__ unsigned pk2(float a, float b) {
    union { __bf16 h; unsigned short s; } ua, ub;
    ua.h = (__bf16)a; ub.h = (__bf16)b;
    return (unsigned)ua.s | ((unsigned)ub.s << 16);
}

// convert 16 f32 -> 16 bf16 and store 32B to LDS (two ds_store_b128)
__device__ __forceinline__ void cvt_store32(unsigned char* d, const float* f) {
    u32x4 q0 = { pk2(f[0], f[1]),  pk2(f[2], f[3]),  pk2(f[4], f[5]),   pk2(f[6], f[7]) };
    u32x4 q1 = { pk2(f[8], f[9]),  pk2(f[10], f[11]), pk2(f[12], f[13]), pk2(f[14], f[15]) };
    *(u32x4*)(d)      = q0;
    *(u32x4*)(d + 16) = q1;
}

// async global -> LDS copy of 32 bytes (two B128 issues; the `offset:` imm is
// added to BOTH the LDS and global addresses per ISA 08_async_tensor 4.4)
__device__ __forceinline__ void async_copy32(unsigned lds_off, const void* gptr) {
    asm volatile("global_load_async_to_lds_b128 %0, %1, off\n\t"
                 "global_load_async_to_lds_b128 %0, %1, off offset:16"
                 :: "v"(lds_off), "v"(gptr) : "memory");
}
__device__ __forceinline__ void wait_async0() {
    asm volatile("s_wait_asynccnt 0x0" ::: "memory");
}

// =====================================================================
// init: zero per-expert counters
// =====================================================================
__global__ void moe_init(int* __restrict__ counts) {
    if (threadIdx.x < NEXP) counts[threadIdx.x] = 0;
}

// =====================================================================
// router: logits -> group-limited greedy top-k -> per-expert token lists
// one wave (32 lanes) per token
// =====================================================================
__global__ __launch_bounds__(256)
void moe_router(const float* __restrict__ x, const float* __restrict__ gw,
                int* __restrict__ counts, int* __restrict__ idx,
                float* __restrict__ wgt) {
    const int lane = threadIdx.x & 31;
    const int wv   = threadIdx.x >> 5;
    const int t    = blockIdx.x * 8 + wv;

    float acc[NEXP];
#pragma unroll
    for (int e = 0; e < NEXP; e++) acc[e] = 0.0f;

    const float* xr = x + (size_t)t * HDIM;
    for (int h = lane; h < HDIM; h += 32) {
        float xv = xr[h];
#pragma unroll
        for (int e = 0; e < NEXP; e++) acc[e] += xv * gw[e * HDIM + h];
    }
#pragma unroll
    for (int e = 0; e < NEXP; e++)
        for (int off = 16; off; off >>= 1) acc[e] += __shfl_xor(acc[e], off, 32);

    if (lane == 0) {
        float gmax[4];
#pragma unroll
        for (int g = 0; g < 4; g++) {
            float m = acc[4 * g];
            for (int j = 1; j < 4; j++) m = fmaxf(m, acc[4 * g + j]);
            gmax[g] = m;
        }
        int ga = 0;
        for (int g = 1; g < 4; g++) if (gmax[g] > gmax[ga]) ga = g;
        int gb = -1;
        for (int g = 0; g < 4; g++) {
            if (g == ga) continue;
            if (gb < 0 || gmax[g] > gmax[gb]) gb = g;
        }
        float l[NEXP];
#pragma unroll
        for (int e = 0; e < NEXP; e++) {
            int g = e >> 2;
            l[e] = (g == ga || g == gb) ? acc[e] : 0.0f;
        }
        float mx = l[0];
        for (int e = 1; e < NEXP; e++) mx = fmaxf(mx, l[e]);
        float p[NEXP], s = 0.0f;
#pragma unroll
        for (int e = 0; e < NEXP; e++) { p[e] = __expf(l[e] - mx); s += p[e]; }
        float inv = 1.0f / s;
#pragma unroll
        for (int e = 0; e < NEXP; e++) p[e] *= inv;
        for (int k = 0; k < 4; k++) {
            int be = 0;
            for (int e = 1; e < NEXP; e++) if (p[e] > p[be]) be = e;
            float w = p[be] * RSCALE;
            int pos = atomicAdd(&counts[be], 1);
            idx[be * NTOK + pos] = t;
            wgt[be * NTOK + pos] = w;
            p[be] = -1.0f;
        }
    }
}

// =====================================================================
// GEMM1: act[r, 0:Icols] = silu(xg @ Wg) * (xg @ Wu)   (bf16 out)
// software-pipelined: double-buffered LDS, one barrier per K-step;
// next tile's global loads overlap current tile's WMMAs.
// grid: (NTOK/128, Icols/128), block 256 (8 wave32)
// =====================================================================
__global__ __launch_bounds__(256)
void moe_gemm1(const float* __restrict__ x, const float* __restrict__ wgu,
               __bf16* __restrict__ act, const int* __restrict__ idxp,
               const int* __restrict__ cntp, int Icols, int upOff, int ldw) {
    const int cnt = cntp ? *cntp : NTOK;
    const int tm0 = blockIdx.x * 128;
    if (tm0 >= cnt) return;
    const int tn0 = blockIdx.y * 128;

    __shared__ alignas(16) unsigned char smem[2 * 3 * TILE_BYTES];

    const int tid  = threadIdx.x;
    const int lane = tid & 31;
    const int wv   = tid >> 5;
    const int wm   = wv & 3;    // m offset * 32
    const int wn   = wv >> 2;   // n offset * 64
    const int hl   = lane >> 4;
    const int l16  = lane & 15;

    v8f accg[2][4], accu[2][4];
#pragma unroll
    for (int mi = 0; mi < 2; mi++)
#pragma unroll
        for (int ni = 0; ni < 4; ni++)
#pragma unroll
            for (int r = 0; r < 8; r++) { accg[mi][ni][r] = 0.f; accu[mi][ni][r] = 0.f; }

    // ---- writer addressing ----
    const int am  = tid >> 1;            // 0..127 (m row)
    const int aks = (tid & 1) * 16;      // k segment 0 / 16
    const int gm  = tm0 + am;
    const int grow = idxp ? ((gm < cnt) ? idxp[gm] : 0) : gm;
    const float* aSrc = x + (size_t)grow * HDIM + aks;
    const unsigned aoff = am * LROW + aks * 2;

    const int bn  = tid & 127;           // n row
    const int bks = (tid >> 7) * 16;     // k segment
    const float* gSrc = wgu + (size_t)bks * ldw + (tn0 + bn);
    const float* uSrc = wgu + (size_t)bks * ldw + (upOff + tn0 + bn);
    const unsigned boff = bn * LROW + bks * 2;

    R16 ra, rg, ru;
    auto load_tile = [&](int k) {
        const float* s = aSrc + k;
        ra.v[0] = *(const f32x4*)(s);
        ra.v[1] = *(const f32x4*)(s + 4);
        ra.v[2] = *(const f32x4*)(s + 8);
        ra.v[3] = *(const f32x4*)(s + 12);
        const float* g = gSrc + (size_t)k * ldw;
        const float* u = uSrc + (size_t)k * ldw;
#pragma unroll
        for (int j = 0; j < 16; j++) {
            rg.f[j] = g[(size_t)j * ldw];
            ru.f[j] = u[(size_t)j * ldw];
        }
    };
    auto store_tile = [&](unsigned char* base) {
        cvt_store32(base + aoff, ra.f);
        cvt_store32(base + TILE_BYTES + boff, rg.f);
        cvt_store32(base + 2 * TILE_BYTES + boff, ru.f);
    };

    load_tile(0);
    store_tile(smem);
    __syncthreads();

    int buf = 0;
    for (int k0 = 0; k0 < HDIM; k0 += 32) {
        const unsigned char* cur = smem + buf * (3 * TILE_BYTES);
        unsigned char* nxt = smem + (buf ^ 1) * (3 * TILE_BYTES);
        const bool has_next = (k0 + 32) < HDIM;
        if (has_next) load_tile(k0 + 32);   // global latency overlaps WMMAs below

        FragB16 fa[2], fg[4], fu[4];
#pragma unroll
        for (int mi = 0; mi < 2; mi++) {
            const unsigned char* p = cur + (wm * 32 + mi * 16 + l16) * LROW + hl * 32;
            fa[mi].q[0] = *(const u32x4*)(p);
            fa[mi].q[1] = *(const u32x4*)(p + 16);
        }
#pragma unroll
        for (int ni = 0; ni < 4; ni++) {
            const unsigned char* pg = cur + TILE_BYTES + (wn * 64 + ni * 16 + l16) * LROW + hl * 32;
            fg[ni].q[0] = *(const u32x4*)(pg);
            fg[ni].q[1] = *(const u32x4*)(pg + 16);
            const unsigned char* pu = cur + 2 * TILE_BYTES + (wn * 64 + ni * 16 + l16) * LROW + hl * 32;
            fu[ni].q[0] = *(const u32x4*)(pu);
            fu[ni].q[1] = *(const u32x4*)(pu + 16);
        }
#pragma unroll
        for (int mi = 0; mi < 2; mi++)
#pragma unroll
            for (int ni = 0; ni < 4; ni++) {
                accg[mi][ni] = __builtin_amdgcn_wmma_f32_16x16x32_bf16(
                    false, fa[mi].v, false, fg[ni].v, (short)0, accg[mi][ni], false, false);
                accu[mi][ni] = __builtin_amdgcn_wmma_f32_16x16x32_bf16(
                    false, fa[mi].v, false, fu[ni].v, (short)0, accu[mi][ni], false, false);
            }

        if (has_next) store_tile(nxt);
        __syncthreads();
        buf ^= 1;
    }

    // epilogue: silu(g)*u -> bf16 act  (fast rcp instead of IEEE div chain)
#pragma unroll
    for (int mi = 0; mi < 2; mi++)
#pragma unroll
        for (int ni = 0; ni < 4; ni++) {
            const int gn = tn0 + wn * 64 + ni * 16 + l16;
            const int mb = tm0 + wm * 32 + mi * 16 + hl * 8;
#pragma unroll
            for (int r = 0; r < 8; r++) {
                float g = accg[mi][ni][r];
                float u = accu[mi][ni][r];
                float v = g * __builtin_amdgcn_rcpf(1.0f + __expf(-g)) * u;
                act[(size_t)(mb + r) * Icols + gn] = (__bf16)v;
            }
        }
}

// =====================================================================
// GEMM2: y = act @ Wd ; scatter out[idx[r], :] (+)= wgt[r] * y[r, :]
// A (bf16) staged via async global->LDS (ASYNCcnt); B via regs with cvt.
// grid: (NTOK/128, HDIM/128), block 256
// =====================================================================
__global__ __launch_bounds__(256)
void moe_gemm2(const __bf16* __restrict__ act, const float* __restrict__ wd,
               float* __restrict__ out, const int* __restrict__ idxp,
               const float* __restrict__ wgtp, const int* __restrict__ cntp,
               int Kdim, int accum) {
    const int cnt = cntp ? *cntp : NTOK;
    const int tm0 = blockIdx.x * 128;
    if (tm0 >= cnt) return;
    const int tn0 = blockIdx.y * 128;

    __shared__ alignas(16) unsigned char smem[2 * 2 * TILE_BYTES];
    const unsigned lds0 = (unsigned)(size_t)(void*)smem;  // LDS byte offset (addr[31:0])

    const int tid  = threadIdx.x;
    const int lane = tid & 31;
    const int wv   = tid >> 5;
    const int wm   = wv & 3;
    const int wn   = wv >> 2;
    const int hl   = lane >> 4;
    const int l16  = lane & 15;

    v8f acc[2][4];
#pragma unroll
    for (int mi = 0; mi < 2; mi++)
#pragma unroll
        for (int ni = 0; ni < 4; ni++)
#pragma unroll
            for (int r = 0; r < 8; r++) acc[mi][ni][r] = 0.f;

    const int am  = tid >> 1;
    const int aks = (tid & 1) * 16;
    const unsigned char* aSrcG =
        (const unsigned char*)(act + (size_t)(tm0 + am) * Kdim) + aks * 2;
    const unsigned aoff = am * LROW + aks * 2;

    const int bn  = tid & 127;
    const int bks = (tid >> 7) * 16;
    const float* bSrc = wd + (size_t)bks * HDIM + (tn0 + bn);
    const unsigned boff = bn * LROW + bks * 2;

    R16 rb;
    auto loadB = [&](int k) {
        const float* s = bSrc + (size_t)k * HDIM;
#pragma unroll
        for (int j = 0; j < 16; j++) rb.f[j] = s[(size_t)j * HDIM];
    };
    auto asyncA = [&](int k, int b) {
        async_copy32(lds0 + (unsigned)b * (2 * TILE_BYTES) + aoff, aSrcG + (size_t)k * 2);
    };
    auto storeB = [&](int b) {
        cvt_store32(smem + b * (2 * TILE_BYTES) + TILE_BYTES + boff, rb.f);
    };

    asyncA(0, 0);
    loadB(0);
    storeB(0);
    wait_async0();
    __syncthreads();

    int buf = 0;
    for (int k0 = 0; k0 < Kdim; k0 += 32) {
        const unsigned char* cur = smem + buf * (2 * TILE_BYTES);
        const bool has_next = (k0 + 32) < Kdim;
        if (has_next) {
            asyncA(k0 + 32, buf ^ 1);   // direct global->LDS, no VGPR round-trip
            loadB(k0 + 32);
        }

        FragB16 fa[2], fb[4];
#pragma unroll
        for (int mi = 0; mi < 2; mi++) {
            const unsigned char* p = cur + (wm * 32 + mi * 16 + l16) * LROW + hl * 32;
            fa[mi].q[0] = *(const u32x4*)(p);
            fa[mi].q[1] = *(const u32x4*)(p + 16);
        }
#pragma unroll
        for (int ni = 0; ni < 4; ni++) {
            const unsigned char* p = cur + TILE_BYTES + (wn * 64 + ni * 16 + l16) * LROW + hl * 32;
            fb[ni].q[0] = *(const u32x4*)(p);
            fb[ni].q[1] = *(const u32x4*)(p + 16);
        }
#pragma unroll
        for (int mi = 0; mi < 2; mi++)
#pragma unroll
            for (int ni = 0; ni < 4; ni++)
                acc[mi][ni] = __builtin_amdgcn_wmma_f32_16x16x32_bf16(
                    false, fa[mi].v, false, fb[ni].v, (short)0, acc[mi][ni], false, false);

        if (has_next) storeB(buf ^ 1);
        wait_async0();
        __syncthreads();
        buf ^= 1;
    }

    // epilogue: weighted scatter (race-free: kernels serialize on stream,
    // each token appears at most once per expert list)
#pragma unroll
    for (int mi = 0; mi < 2; mi++)
#pragma unroll
        for (int ni = 0; ni < 4; ni++) {
            const int gn = tn0 + wn * 64 + ni * 16 + l16;
            const int mb = tm0 + wm * 32 + mi * 16 + hl * 8;
#pragma unroll
            for (int r = 0; r < 8; r++) {
                const int lm = mb + r;
                if (lm < cnt) {
                    const int   row = idxp ? idxp[lm] : lm;
                    const float w   = wgtp ? wgtp[lm] : 1.0f;
                    const size_t o  = (size_t)row * HDIM + gn;
                    const float  v  = acc[mi][ni][r] * w;
                    if (accum) out[o] += v;
                    else       out[o]  = v;
                }
            }
        }
}

// =====================================================================
// host launcher
// =====================================================================
extern "C" void kernel_launch(void* const* d_in, const int* in_sizes, int n_in,
                              void* d_out, int out_size, void* d_ws, size_t ws_size,
                              hipStream_t stream) {
    (void)in_sizes; (void)n_in; (void)out_size; (void)ws_size;

    const float* x   = (const float*)d_in[0];  // [T, H]
    const float* gw  = (const float*)d_in[1];  // [E, H]
    const float* wgu = (const float*)d_in[2];  // [E, H, 2I]
    const float* wd  = (const float*)d_in[3];  // [E, I, H]
    const float* sgu = (const float*)d_in[4];  // [H, 2*SI]
    const float* sd  = (const float*)d_in[5];  // [SI, H]
    float* out = (float*)d_out;

    // workspace layout
    char* ws = (char*)d_ws;
    int*    counts = (int*)ws;                                    // 16 ints
    int*    idx    = (int*)(ws + 256);                            // E*T ints
    float*  wgt    = (float*)(ws + 256 + NEXP * NTOK * 4);        // E*T floats
    __bf16* act    = (__bf16*)(ws + 262400);                      // T*SI bf16 max

    moe_init<<<1, 32, 0, stream>>>(counts);
    moe_router<<<NTOK / 8, 256, 0, stream>>>(x, gw, counts, idx, wgt);

    // shared expert first: initializes d_out (accum = 0)
    moe_gemm1<<<dim3(NTOK / 128, SIDIM / 128), 256, 0, stream>>>(
        x, sgu, act, nullptr, nullptr, SIDIM, SIDIM, 2 * SIDIM);
    moe_gemm2<<<dim3(NTOK / 128, HDIM / 128), 256, 0, stream>>>(
        act, sd, out, nullptr, nullptr, nullptr, SIDIM, 0);

    // routed experts: sparse (only count[e] gathered rows do real work)
    for (int e = 0; e < NEXP; e++) {
        moe_gemm1<<<dim3(NTOK / 128, IDIM / 128), 256, 0, stream>>>(
            x, wgu + (size_t)e * HDIM * 2 * IDIM, act,
            idx + e * NTOK, counts + e, IDIM, IDIM, 2 * IDIM);
        moe_gemm2<<<dim3(NTOK / 128, HDIM / 128), 256, 0, stream>>>(
            act, wd + (size_t)e * IDIM * HDIM, out,
            idx + e * NTOK, wgt + e * NTOK, counts + e, IDIM, 1);
    }
}
